// EncoderBlock_58634893525346
// MI455X (gfx1250) — matrix operand order repaired and checked by reading, not compiled, not run on hardware
//
#include <hip/hip_runtime.h>
#include <hip/hip_bf16.h>

// ---------------------------------------------------------------------------
// Types for CDNA5 WMMA (wave32): v_wmma_f32_16x16x32_bf16
// ---------------------------------------------------------------------------
typedef __bf16 bf16_t;
typedef __attribute__((ext_vector_type(16))) __bf16 v16bf;
typedef __attribute__((ext_vector_type(8)))  __bf16 bf16x8;
typedef __attribute__((ext_vector_type(8)))  float  v8f;

union BF16x16 { v16bf v; bf16x8 h[2]; };

#define WMMA_BF16(a, b, c) \
  __builtin_amdgcn_wmma_f32_16x16x32_bf16(false, (a), false, (b), (short)0, (c), false, false)

constexpr int S_LEN   = 1024;
constexpr int DMODEL  = 1024;
constexpr int NHEADS  = 16;
constexpr int DK      = 64;
constexpr int BATCH   = 4;
constexpr int MTOK    = BATCH * S_LEN;   // 4096 tokens
constexpr int DFF     = 4096;

// ---------------------------------------------------------------------------
// f32 [K,N] row-major -> bf16 [N,K] row-major, LDS-tiled 32x32 transpose:
// coalesced 128B reads along N, coalesced bf16 writes along K, +1 pad to
// avoid LDS bank conflicts on the transposed read.
// ---------------------------------------------------------------------------
__global__ __launch_bounds__(256) void k_cvt_transpose(const float* __restrict__ in,
                                                       bf16_t* __restrict__ out,
                                                       int K, int N) {
  __shared__ float tile[32][33];
  const int k0 = blockIdx.y * 32;
  const int n0 = blockIdx.x * 32;
  const int tx = threadIdx.x & 31;
  const int ty = threadIdx.x >> 5;       // 8 rows per pass
#pragma unroll
  for (int i = 0; i < 32; i += 8)
    tile[ty + i][tx] = in[(size_t)(k0 + ty + i) * N + n0 + tx];
  __syncthreads();
#pragma unroll
  for (int i = 0; i < 32; i += 8)
    out[(size_t)(n0 + ty + i) * K + k0 + tx] = (bf16_t)tile[tx][ty + i];
}

// ---------------------------------------------------------------------------
// LayerNorm, faithful to reference: mean, UNBIASED var (ddof=1), eps on std.
// One 256-thread block per row of 1024. Output bf16 (GEMM A-operand).
// ---------------------------------------------------------------------------
__global__ __launch_bounds__(256) void k_layernorm(const float* __restrict__ x,
                                                   bf16_t* __restrict__ out,
                                                   const float* __restrict__ alpha,
                                                   const float* __restrict__ beta) {
  __shared__ float red_s[8], red_q[8];
  const int row = blockIdx.x;
  const float* xr = x + (size_t)row * DMODEL;
  float v[4], s = 0.f, sq = 0.f;
#pragma unroll
  for (int i = 0; i < 4; ++i) {
    v[i] = xr[threadIdx.x + i * 256];
    s += v[i];
    sq += v[i] * v[i];
  }
#pragma unroll
  for (int m = 1; m < 32; m <<= 1) {
    s  += __shfl_xor(s, m, 32);
    sq += __shfl_xor(sq, m, 32);
  }
  const int wave = threadIdx.x >> 5, lane = threadIdx.x & 31;
  if (lane == 0) { red_s[wave] = s; red_q[wave] = sq; }
  __syncthreads();
  float ts = 0.f, tq = 0.f;
#pragma unroll
  for (int i = 0; i < 8; ++i) { ts += red_s[i]; tq += red_q[i]; }
  const float mean = ts * (1.0f / DMODEL);
  const float var  = fmaxf((tq - (float)DMODEL * mean * mean) * (1.0f / (DMODEL - 1)), 0.f);
  const float inv  = alpha[0] / (sqrtf(var) + 1e-6f);
  const float bb   = beta[0];
  bf16_t* orow = out + (size_t)row * DMODEL;
#pragma unroll
  for (int i = 0; i < 4; ++i)
    orow[threadIdx.x + i * 256] = (bf16_t)((v[i] - mean) * inv + bb);
}

// ---------------------------------------------------------------------------
// WMMA bf16 GEMM: C[M,N] = A[M,K] @ B[K,N] + bias, with B given transposed
// (BT[N,K]).  Block = 256 threads = 8 waves arranged 2(M) x 4(N); each wave
// owns a 32x64 tile (8 fp32 WMMA accumulators).  Operands stream straight
// from L2 (192MB holds the whole working set) in WMMA-native lane layout.
// Epilogue modes:
//  0: f32 row-major (+bias)
//  1: f32 row-major (+bias) + residual add
//  2: bf16 row-major, ReLU
//  3: bf16 head-major   [B,H,S,dk]   (Q/K for attention)
//  4: bf16 head-major-T [B,H,dk,S]   (V^T for attention)
// ---------------------------------------------------------------------------
__global__ __launch_bounds__(256) void k_gemm(const bf16_t* __restrict__ A,
                                              const bf16_t* __restrict__ BT,
                                              const float* __restrict__ bias,
                                              int M, int N, int K, int mode,
                                              float* __restrict__ outF,
                                              bf16_t* __restrict__ outB,
                                              const float* __restrict__ resid) {
  const int lane = threadIdx.x & 31;
  const int wave = threadIdx.x >> 5;
  const int hi   = lane >> 4;     // half-wave select
  const int r    = lane & 15;
  const int m0 = blockIdx.y * 64 + (wave >> 2) * 32;
  const int n0 = blockIdx.x * 256 + (wave & 3) * 64;

  v8f acc[2][4] = {};

  const bf16_t* arow0 = A + (size_t)(m0 + r) * K;
  const bf16_t* arow1 = A + (size_t)(m0 + 16 + r) * K;

  for (int k0 = 0; k0 < K; k0 += 32) {
    BF16x16 a0, a1;
    a0.h[0] = *(const bf16x8*)(arow0 + k0 + hi * 8);
    a0.h[1] = *(const bf16x8*)(arow0 + k0 + 16 + hi * 8);
    a1.h[0] = *(const bf16x8*)(arow1 + k0 + hi * 8);
    a1.h[1] = *(const bf16x8*)(arow1 + k0 + 16 + hi * 8);
#pragma unroll
    for (int j = 0; j < 4; ++j) {
      const bf16_t* brow = BT + (size_t)(n0 + j * 16 + r) * K + k0 + hi * 8;
      BF16x16 b;
      b.h[0] = *(const bf16x8*)brow;
      b.h[1] = *(const bf16x8*)(brow + 16);
      acc[0][j] = WMMA_BF16(a0.v, b.v, acc[0][j]);
      acc[1][j] = WMMA_BF16(a1.v, b.v, acc[1][j]);
    }
  }

#pragma unroll
  for (int j = 0; j < 4; ++j) {
    const int col = n0 + j * 16 + r;
    const float bv = bias[col];
#pragma unroll
    for (int i = 0; i < 2; ++i) {
#pragma unroll
      for (int e = 0; e < 8; ++e) {
        const int row = m0 + i * 16 + e + hi * 8;   // C layout: M = e + 8*hi
        float val = acc[i][j][e] + bv;
        if (mode == 0) {
          outF[(size_t)row * N + col] = val;
        } else if (mode == 1) {
          outF[(size_t)row * N + col] = val + resid[(size_t)row * N + col];
        } else if (mode == 2) {
          outB[(size_t)row * N + col] = (bf16_t)fmaxf(val, 0.f);
        } else {
          const int b = row >> 10, s = row & 1023;
          const int hh = col >> 6, d = col & 63;
          if (mode == 3)
            outB[(((size_t)b * NHEADS + hh) * S_LEN + s) * DK + d] = (bf16_t)val;
          else
            outB[(((size_t)b * NHEADS + hh) * DK + d) * S_LEN + s] = (bf16_t)val;
        }
      }
    }
  }
}

// ---------------------------------------------------------------------------
// Flash attention.  Grid = (B*H, 8); block = 256 threads = 8 waves; each wave
// owns exactly ONE 16-row q-tile (4096 waves total for latency hiding).
// Computes S^T = K_blk @ Q^T (both operands load in natural row order), so
// the softmax reduction over keys is 16 in-register values + ONE shfl_xor(16);
// running max/sum are lane scalars (q = lane&15).  Then O^T += V^T_blk @ P^T;
// the P^T B-operand is built from the score C-tiles with a single half-wave
// exchange.  8 WMMAs per 32-key block.
// ---------------------------------------------------------------------------
__global__ __launch_bounds__(256) void k_attention(const bf16_t* __restrict__ Q,
                                                   const bf16_t* __restrict__ Kh,
                                                   const bf16_t* __restrict__ VT,
                                                   const int* __restrict__ mask,
                                                   bf16_t* __restrict__ ctx) {
  __shared__ int smask[S_LEN];
  const int b = blockIdx.x >> 4;
  const int h = blockIdx.x & 15;
  for (int i = threadIdx.x; i < S_LEN; i += 256) smask[i] = mask[b * S_LEN + i];
  __syncthreads();

  const int lane = threadIdx.x & 31, wave = threadIdx.x >> 5;
  const int hi = lane >> 4, r = lane & 15;
  const bf16_t* qbase = Q  + (((size_t)b * NHEADS + h) * S_LEN) * DK;
  const bf16_t* kbase = Kh + (((size_t)b * NHEADS + h) * S_LEN) * DK;
  const bf16_t* vbase = VT + (((size_t)b * NHEADS + h) * DK) * S_LEN;

  const int qt = blockIdx.y * 8 + wave;    // one 16-row q-tile per wave
  const int q0 = qt * 16;

  // Q^T as B-operand, two d-chunks of 32 (lane r = query column)
  BF16x16 bq[2];
#pragma unroll
  for (int c = 0; c < 2; ++c) {
    const bf16_t* p = qbase + (size_t)(q0 + r) * DK + c * 32 + hi * 8;
    bq[c].h[0] = *(const bf16x8*)p;
    bq[c].h[1] = *(const bf16x8*)(p + 16);
  }

  v8f ot[4] = {};
  float m_run = -3.0e38f, l_run = 0.f;

  for (int key0 = 0; key0 < S_LEN; key0 += 32) {
    // ---- S^T tiles: [16 keys x 16 q] x2 -------------------------------
    v8f st0 = {}, st1 = {};
#pragma unroll
    for (int c = 0; c < 2; ++c) {
      BF16x16 ak0, ak1;
      const bf16_t* p0 = kbase + (size_t)(key0 + r) * DK + c * 32;
      ak0.h[0] = *(const bf16x8*)(p0 + hi * 8);
      ak0.h[1] = *(const bf16x8*)(p0 + 16 + hi * 8);
      const bf16_t* p1 = kbase + (size_t)(key0 + 16 + r) * DK + c * 32;
      ak1.h[0] = *(const bf16x8*)(p1 + hi * 8);
      ak1.h[1] = *(const bf16x8*)(p1 + 16 + hi * 8);
      st0 = WMMA_BF16(ak0.v, bq[c].v, st0);
      st1 = WMMA_BF16(ak1.v, bq[c].v, st1);
    }
    // ---- scale + mask + online softmax --------------------------------
    float p0v[8], p1v[8];
    float lm = -3.0e38f;
#pragma unroll
    for (int e = 0; e < 8; ++e) {
      const int ky = key0 + e + hi * 8;       // C layout: key row = e + 8*hi
      float s0 = st0[e] * 0.125f;             // 1/sqrt(64)
      float s1 = st1[e] * 0.125f;
      if (smask[ky] == 0)      s0 = -1.0e9f;
      if (smask[ky + 16] == 0) s1 = -1.0e9f;
      p0v[e] = s0; p1v[e] = s1;
      lm = fmaxf(lm, fmaxf(s0, s1));
    }
    lm = fmaxf(lm, __shfl_xor(lm, 16, 32));
    const float m_new = fmaxf(m_run, lm);
    const float corr = __expf(m_run - m_new);
    float lsum = 0.f;
#pragma unroll
    for (int e = 0; e < 8; ++e) {
      p0v[e] = __expf(p0v[e] - m_new);
      p1v[e] = __expf(p1v[e] - m_new);
      lsum += p0v[e] + p1v[e];
    }
    lsum += __shfl_xor(lsum, 16, 32);
    l_run = l_run * corr + lsum;
    m_run = m_new;
#pragma unroll
    for (int t = 0; t < 4; ++t)
#pragma unroll
      for (int e = 0; e < 8; ++e) ot[t][e] *= corr;

    // ---- build P^T B-operand via one half-wave exchange ---------------
    BF16x16 pB;
#pragma unroll
    for (int e = 0; e < 8; ++e) {
      const float o0 = __shfl_xor(p0v[e], 16, 32);
      const float o1 = __shfl_xor(p1v[e], 16, 32);
      pB.v[e]     = (bf16_t)(hi ? o1 : p0v[e]);  // keys base+0..7 / base+16..23
      pB.v[e + 8] = (bf16_t)(hi ? p1v[e] : o0);  // keys base+8..15 / base+24..31
    }
    // ---- O^T += V^T_blk @ P^T -----------------------------------------
#pragma unroll
    for (int t = 0; t < 4; ++t) {
      BF16x16 av;
      const bf16_t* p = vbase + (size_t)(t * 16 + r) * S_LEN + key0;
      av.h[0] = *(const bf16x8*)(p + hi * 16);
      av.h[1] = *(const bf16x8*)(p + hi * 16 + 8);
      ot[t] = WMMA_BF16(av.v, pB.v, ot[t]);
    }
  }

  const float inv_l = 1.0f / l_run;
  bf16_t* crow = ctx + ((size_t)b * S_LEN + q0 + r) * DMODEL + h * DK;
#pragma unroll
  for (int t = 0; t < 4; ++t)
#pragma unroll
    for (int e = 0; e < 8; ++e)
      crow[t * 16 + e + hi * 8] = (bf16_t)(ot[t][e] * inv_l);
}

// ---------------------------------------------------------------------------
// Host-side orchestration (graph-capture safe: launches only, all on stream)
// ---------------------------------------------------------------------------
extern "C" void kernel_launch(void* const* d_in, const int* in_sizes, int n_in,
                              void* d_out, int out_size, void* d_ws, size_t ws_size,
                              hipStream_t stream) {
  (void)in_sizes; (void)n_in; (void)out_size; (void)ws_size;
  const float* x    = (const float*)d_in[0];
  const int*   mask = (const int*)d_in[1];
  const float* wq = (const float*)d_in[2];  const float* bq = (const float*)d_in[3];
  const float* wk = (const float*)d_in[4];  const float* bk = (const float*)d_in[5];
  const float* wv = (const float*)d_in[6];  const float* bv = (const float*)d_in[7];
  const float* wo = (const float*)d_in[8];  const float* bo = (const float*)d_in[9];
  const float* w1 = (const float*)d_in[10]; const float* b1 = (const float*)d_in[11];
  const float* w2 = (const float*)d_in[12]; const float* b2 = (const float*)d_in[13];
  const float* alpha1 = (const float*)d_in[14]; const float* beta1 = (const float*)d_in[15];
  const float* alpha2 = (const float*)d_in[16]; const float* beta2 = (const float*)d_in[17];
  float* out = (float*)d_out;

  char* ws = (char*)d_ws;
  size_t off = 0;
  auto alloc = [&](size_t bytes) -> void* {
    void* p = ws + off;
    off += (bytes + 255) & ~(size_t)255;
    return p;
  };
  bf16_t* wqT = (bf16_t*)alloc((size_t)DMODEL * DMODEL * 2);
  bf16_t* wkT = (bf16_t*)alloc((size_t)DMODEL * DMODEL * 2);
  bf16_t* wvT = (bf16_t*)alloc((size_t)DMODEL * DMODEL * 2);
  bf16_t* woT = (bf16_t*)alloc((size_t)DMODEL * DMODEL * 2);
  bf16_t* w1T = (bf16_t*)alloc((size_t)DMODEL * DFF * 2);
  bf16_t* w2T = (bf16_t*)alloc((size_t)DFF * DMODEL * 2);
  bf16_t* xn  = (bf16_t*)alloc((size_t)MTOK * DMODEL * 2);   // reused for LN1 & LN2
  bf16_t* qh  = (bf16_t*)alloc((size_t)MTOK * DMODEL * 2);
  bf16_t* kh  = (bf16_t*)alloc((size_t)MTOK * DMODEL * 2);
  bf16_t* vT  = (bf16_t*)alloc((size_t)MTOK * DMODEL * 2);
  bf16_t* ctx = (bf16_t*)alloc((size_t)MTOK * DMODEL * 2);
  float*  x1  = (float*) alloc((size_t)MTOK * DMODEL * 4);
  bf16_t* h1  = (bf16_t*)alloc((size_t)MTOK * DFF * 2);

  // 1) weights -> bf16 transposed (B^T layout for WMMA), LDS-tiled
  {
    dim3 gSq(DMODEL / 32, DMODEL / 32);
    k_cvt_transpose<<<gSq, 256, 0, stream>>>(wq, wqT, DMODEL, DMODEL);
    k_cvt_transpose<<<gSq, 256, 0, stream>>>(wk, wkT, DMODEL, DMODEL);
    k_cvt_transpose<<<gSq, 256, 0, stream>>>(wv, wvT, DMODEL, DMODEL);
    k_cvt_transpose<<<gSq, 256, 0, stream>>>(wo, woT, DMODEL, DMODEL);
    dim3 g1T(DFF / 32, DMODEL / 32);     // w1 [1024,4096] -> [4096,1024]
    k_cvt_transpose<<<g1T, 256, 0, stream>>>(w1, w1T, DMODEL, DFF);
    dim3 g2T(DMODEL / 32, DFF / 32);     // w2 [4096,1024] -> [1024,4096]
    k_cvt_transpose<<<g2T, 256, 0, stream>>>(w2, w2T, DFF, DMODEL);
  }

  // 2) LN1
  k_layernorm<<<MTOK, 256, 0, stream>>>(x, xn, alpha1, beta1);

  // 3) Q,K,V projections (WMMA), epilogues write attention-native layouts
  dim3 gQ(DMODEL / 256, MTOK / 64);
  k_gemm<<<gQ, 256, 0, stream>>>(xn, wqT, bq, MTOK, DMODEL, DMODEL, 3, nullptr, qh, nullptr);
  k_gemm<<<gQ, 256, 0, stream>>>(xn, wkT, bk, MTOK, DMODEL, DMODEL, 3, nullptr, kh, nullptr);
  k_gemm<<<gQ, 256, 0, stream>>>(xn, wvT, bv, MTOK, DMODEL, DMODEL, 4, nullptr, vT, nullptr);

  // 4) flash attention: one q-tile per wave, 4096 waves
  dim3 gA(BATCH * NHEADS, S_LEN / 16 / 8);
  k_attention<<<gA, 256, 0, stream>>>(qh, kh, vT, mask, ctx);

  // 5) output projection + residual (x1 = x + ctx@wo + bo)
  k_gemm<<<gQ, 256, 0, stream>>>(ctx, woT, bo, MTOK, DMODEL, DMODEL, 1, x1, nullptr, x);

  // 6) LN2
  k_layernorm<<<MTOK, 256, 0, stream>>>(x1, xn, alpha2, beta2);

  // 7) FFN1 with fused ReLU -> bf16
  dim3 gF1(DFF / 256, MTOK / 64);
  k_gemm<<<gF1, 256, 0, stream>>>(xn, w1T, b1, MTOK, DFF, DMODEL, 2, nullptr, h1, nullptr);

  // 8) FFN2 + residual -> d_out
  k_gemm<<<gQ, 256, 0, stream>>>(h1, w2T, b2, MTOK, DMODEL, DFF, 1, out, nullptr, x1);
}
